// VocalOscillator_77025943486903
// MI455X (gfx1250) — matrix operand order repaired and checked by reading, not compiled, not run on hardware
//
#include <hip/hip_runtime.h>
#include <hip/hip_bf16.h>
#include <stdint.h>

// ---------------- problem constants ----------------
constexpr int   B_    = 4;
constexpr int   T_    = 64000;
constexpr int   H_    = 96;
constexpr float FS_   = 44100.0f;
constexpr float TWO_PI_F = 6.28318530717958647692f;
constexpr float HALF_PI_F = 1.57079632679489661923f;
constexpr float INV_TWO_PI_F = 0.15915494309189533577f;

// FIR formant filter (replaces rfft * Lorentzian * irfft):
constexpr int   KHALF  = 1032;              // half length (center tap)
constexpr int   LH     = 2065;              // impulse response length (2*KHALF+1)
constexpr int   NCHUNK = 65;                // WMMA K chunks of 32  (Kred = 2080 = LH + 15)
constexpr int   PADLEN = T_ + 2336;         // circularly padded f16 signal per batch
constexpr int   TILE   = 256;               // outputs per WMMA wave (16x16 D tile)

// WMMA conv blocking: 10 waves/block, 10 tiles/block, 25 blocks/batch (exact)
constexpr int   TPB_CONV   = 10;                         // tiles (=waves) per block
constexpr int   BLK_PER_B  = (T_ / TILE) / TPB_CONV;     // 25
constexpr int   SA_HALVES  = NCHUNK * 32 * 16;           // 33280 (66,560 B)
constexpr int   SX_HALVES  = 4640;                       // (TPB-1)*256 + 2320 -> rounded to 32

typedef _Float16 v16h __attribute__((ext_vector_type(16)));
typedef float    v8f  __attribute__((ext_vector_type(8)));
typedef unsigned int u32x4 __attribute__((ext_vector_type(4)));
typedef int          i32x4 __attribute__((ext_vector_type(4)));
typedef int          i32x8 __attribute__((ext_vector_type(8)));

// ---------------------------------------------------------------------------
// Tensor Data Mover: 1D (single-row tile) load of n 16-bit elements into LDS.
// D# built per CDNA5 ISA (group0: count=1, lds_addr,
// 57-bit global addr, type=2; group1: data_size=2B, tile 1 x n).
// ---------------------------------------------------------------------------
__device__ __forceinline__ void tdm_load_1d(uint32_t lds_byte_addr,
                                            const void* gptr,
                                            uint32_t n_elem16) {
#if __HIP_DEVICE_COMPILE__
  const uint64_t ga = (uint64_t)(uintptr_t)gptr;
  u32x4 g0;
  g0.x = 1u;                                             // count=1 (valid), user mode
  g0.y = lds_byte_addr;                                  // lds_addr [63:32]
  g0.z = (uint32_t)(ga & 0xffffffffu);                   // global_addr[31:0]
  g0.w = (uint32_t)((ga >> 32) & 0x01ffffffu) | (2u << 30); // addr[56:32] | type=2
  i32x8 g1;
  g1[0] = (int)(1u << 16);                               // wg_mask=0, data_size=1 (2 bytes)
  g1[1] = (int)((n_elem16 & 0xffffu) << 16);             // tensor_dim0[15:0]
  g1[2] = (int)(((n_elem16 >> 16) & 0xffffu) | (1u << 16)); // tensor_dim0[31:16], tensor_dim1=1
  g1[3] = (int)((n_elem16 & 0xffffu) << 16);             // tile_dim0 = n
  g1[4] = 1;                                             // tile_dim1 = 1, tile_dim2 = 0
  g1[5] = (int)n_elem16;                                 // tensor_dim0_stride[31:0]
  g1[6] = 0;                                             // stride hi / dim1_stride lo
  g1[7] = 0;
  const i32x4 gz = {0, 0, 0, 0};
#if defined(__clang_major__) && (__clang_major__ >= 23)
  const i32x8 gz8 = {0, 0, 0, 0, 0, 0, 0, 0};
  __builtin_amdgcn_tensor_load_to_lds(g0, g1, gz, gz, gz8, 0);
#else
  __builtin_amdgcn_tensor_load_to_lds(g0, g1, gz, gz, 0);
#endif
#else
  (void)lds_byte_addr; (void)gptr; (void)n_elem16;
#endif
}

// ---------------------------------------------------------------------------
// 1) Vibrato + phase integration (inclusive LDS scan over T per batch with
//    serial carry). Also writes f0_vib and the final-phase output.
// ---------------------------------------------------------------------------
__global__ void vox_phase_scan(const float* __restrict__ f0,
                               const float* __restrict__ vr,
                               const float* __restrict__ vd,
                               const float* __restrict__ initp,
                               float* __restrict__ fund,
                               float* __restrict__ f0v,
                               float* __restrict__ finph) {
  const int b   = blockIdx.x;
  const int tid = threadIdx.x;
  __shared__ float s[1024];
  const float init = initp[b];
  const size_t base = (size_t)b * T_;
  float carry = 0.0f;
  const float LN2_OVER_12 = 0.05776226504666210911f; // ln(2)/12
  for (int c0 = 0; c0 < T_; c0 += 1024) {
    const int t = c0 + tid;
    float d = 0.0f;
    if (t < T_) {
      const float f  = f0[base + t];
      const float tt = (float)t * (1.0f / FS_);
      const float vm = __expf(__sinf(TWO_PI_F * vr[base + t] * tt) *
                              (LN2_OVER_12 * vd[base + t]));
      const float fv = (f > 0.0f) ? f * vm : f;
      f0v[base + t]  = fv;
      d = TWO_PI_F * fv * (1.0f / FS_);
    }
    s[tid] = d;
    __syncthreads();
    for (int off = 1; off < 1024; off <<= 1) {
      const float a = (tid >= off) ? s[tid - off] : 0.0f;
      __syncthreads();
      s[tid] += a;
      __syncthreads();
    }
    if (t < T_) fund[base + t] = carry + s[tid] + init;
    carry += s[1023];
    __syncthreads();
  }
  if (tid == 0) finph[b] = fmodf(carry + init, TWO_PI_F);
}

// ---------------------------------------------------------------------------
// 2) Harmonic glottal-flow synthesis: one thread per (b,t); single coalesced
//    pass over the 98 MB amplitude tensor (bandwidth floor of the problem).
//    The reference's phase_coherence blend is algebraically the identity.
// ---------------------------------------------------------------------------
__global__ void vox_harmonics(const float* __restrict__ amp,
                              const float* __restrict__ f0v,
                              const float* __restrict__ fund,
                              const float* __restrict__ oqp,
                              float* __restrict__ sig) {
  const size_t idx = (size_t)blockIdx.x * blockDim.x + threadIdx.x;
  if (idx >= (size_t)B_ * T_) return;
  const float* ap = amp + idx * (size_t)H_;
  __builtin_prefetch(ap, 0, 1);                 // global_prefetch_b8
  const float ph0 = fund[idx];
  const float fv  = f0v[idx];
  const float oq  = oqp[idx];
  const float inv_oq   = 1.0f / oq;
  const float inv_1moq = 1.0f / (1.0f - oq);
  const float nyq_n    = (0.5f * FS_) / fv;
  float acc = 0.0f;
#pragma unroll 4
  for (int h = 0; h < H_; ++h) {
    const float n = (float)(h + 1);
    const float a = (n < nyq_n) ? ap[h] : 0.0f;
    float np = ph0 * n * INV_TWO_PI_F;
    np -= floorf(np);
    float flow;
    if (np < oq) { const float sv = __sinf(np * inv_oq * HALF_PI_F);          flow = sv * sv; }
    else         { const float cv = __cosf((np - oq) * inv_1moq * HALF_PI_F); flow = cv * cv; }
    acc = __fmaf_rn(flow - 0.5f, a, acc);
  }
  sig[idx] = acc;
}

// ---------------------------------------------------------------------------
// 3) Formant impulse response per batch (time-domain Lorentzian pair).
// ---------------------------------------------------------------------------
__global__ void vox_hbuild(const float* __restrict__ ff,
                           const float* __restrict__ fb,
                           const float* __restrict__ fg,
                           float* __restrict__ hbuf) {
  const int b   = blockIdx.x;
  const int mid = T_ / 2;
  float fc[4], bw[4], g[4];
  for (int i = 0; i < 4; ++i) {
    const size_t ix = ((size_t)b * T_ + mid) * 4 + i;
    fc[i] = ff[ix]; bw[i] = fb[ix]; g[i] = fg[ix];
  }
  const float PI_F = 3.14159265358979323846f;
  for (int j = threadIdx.x; j < LH; j += blockDim.x) {
    const float d = (float)(j - KHALF) * (1.0f / FS_);
    float acc = 0.0f;
    for (int i = 0; i < 4; ++i) {
      acc += g[i] * (PI_F * bw[i] / FS_) *
             __expf(-PI_F * bw[i] * fabsf(d)) * __cosf(TWO_PI_F * fc[i] * d);
    }
    hbuf[b * LH + j] = acc;
  }
}

// ---------------------------------------------------------------------------
// 4) Circularly padded f16 signal: xh[b][i] = (f16)sig[(i-KHALF) mod T]
// ---------------------------------------------------------------------------
__global__ void vox_pad16(const float* __restrict__ sig, _Float16* __restrict__ xh) {
  const size_t idx = (size_t)blockIdx.x * blockDim.x + threadIdx.x;
  if (idx >= (size_t)B_ * PADLEN) return;
  const int b = (int)(idx / PADLEN);
  const int i = (int)(idx % PADLEN);
  const int src = (i + T_ - KHALF) % T_;
  xh[idx] = (_Float16)sig[(size_t)b * T_ + src];
}

// ---------------------------------------------------------------------------
// 5) Pack the Toeplitz band of h into a 16-bit A-matrix (16x32) WMMA
//    register layout (contiguous k, same order as B) per batch/chunk: lane L (m=L&15) element e holds
//    K = e + 16*(L>>4);  A[m][j'] = h[j'-m] (0 <= j'-m < LH).
// ---------------------------------------------------------------------------
__global__ void vox_afrag(const float* __restrict__ hbuf, _Float16* __restrict__ afrag) {
  const int bc = blockIdx.x;
  const int b  = bc / NCHUNK;
  const int c  = bc % NCHUNK;
  const int lane = threadIdx.x;
  const int m  = lane & 15;
  const int hi = lane >> 4;
  _Float16* out = afrag + (((size_t)b * NCHUNK + c) * 32 + lane) * 16;
  const float* hb = hbuf + b * LH;
  for (int e = 0; e < 16; ++e) {
    const int klocal = e + hi * 16;
    const int jp   = c * 32 + klocal;
    const int hidx = jp - m;
    const float v  = (hidx >= 0 && hidx < LH) ? hb[hidx] : 0.0f;
    out[e] = (_Float16)v;
  }
}

// ---------------------------------------------------------------------------
// 6) Formant FIR as WMMA matmul, TDM-staged through LDS.
//    Block = 10 waves = 10 consecutive 256-output tiles of one batch.
//    Wave 0 TDM-loads (a) the batch's A-fragment table and (b) the block's
//    shared signal window into LDS, waits TENSORcnt, then __syncthreads.
//    Each wave: 65 x v_wmma_f32_16x16x32_f16 fed by ds_load_b128.
// ---------------------------------------------------------------------------
__global__ void vox_conv_wmma(const _Float16* __restrict__ xh,
                              const _Float16* __restrict__ afrag,
                              float* __restrict__ outf) {
  __shared__ __align__(64) _Float16 sA[SA_HALVES];
  __shared__ __align__(64) _Float16 sX[SX_HALVES];
  const int b   = blockIdx.x / BLK_PER_B;
  const int blk = blockIdx.x % BLK_PER_B;
  const int t0b = blk * (TPB_CONV * TILE);
  const int wave = threadIdx.x >> 5;
  const int lane = threadIdx.x & 31;

  if (wave == 0) {
    tdm_load_1d((uint32_t)(uintptr_t)(void*)sA,
                afrag + (size_t)b * SA_HALVES, (uint32_t)SA_HALVES);
    tdm_load_1d((uint32_t)(uintptr_t)(void*)sX,
                xh + (size_t)b * PADLEN + t0b, (uint32_t)SX_HALVES);
#if __HIP_DEVICE_COMPILE__
    __builtin_amdgcn_s_wait_tensorcnt(0);   // TENSORcnt is per-wave: only issuer waits
#endif
  }
  __syncthreads();

  const int t0l = wave * TILE;
  const int col = lane & 15;
  const int kst = (lane >> 4) * 16;
  const _Float16* xb = sX + t0l + 16 * col + kst;
  const _Float16* ab = sA + lane * 16;
  v8f acc = {};
  for (int ch = 0; ch < NCHUNK; ++ch) {
    const v16h a  = *(const v16h*)(ab + (size_t)ch * 32 * 16);
    const v16h bm = *(const v16h*)(xb + ch * 32);
#if __HIP_DEVICE_COMPILE__
    acc = __builtin_amdgcn_wmma_f32_16x16x32_f16(
        /*neg_a=*/false, a, /*neg_b=*/false, bm,
        /*c_mod=*/(short)0, acc, /*reuse_a=*/false, /*reuse_b=*/false);
#endif
  }
  const int mbase = (lane >> 4) * 8;
  float* op = outf + (size_t)b * T_ + t0b + t0l + 16 * col + mbase;
#pragma unroll
  for (int r = 0; r < 8; ++r) op[r] = acc[r];
}

// ---------------------------------------------------------------------------
// 7) Deterministic standard-normal noise (hash + Box-Muller).
// ---------------------------------------------------------------------------
__global__ void vox_noise(float* __restrict__ nz) {
  const size_t idx = (size_t)blockIdx.x * blockDim.x + threadIdx.x;
  if (idx >= (size_t)B_ * T_) return;
  uint32_t s = (uint32_t)idx * 2654435761u + 0x9E3779B9u;
  s ^= s >> 16; s *= 0x7feb352du; s ^= s >> 15; s *= 0x846ca68bu; s ^= s >> 16;
  uint32_t s2 = (s ^ 0xdeadbeefu) * 0x9E3779B1u;
  s2 ^= s2 >> 16; s2 *= 0x7feb352du; s2 ^= s2 >> 15;
  const float u1 = ((float)s + 1.0f) * (1.0f / 4294967296.0f);
  const float u2 = (float)s2 * (1.0f / 4294967296.0f);
  nz[idx] = sqrtf(-2.0f * __logf(u1)) * __cosf(TWO_PI_F * u2);
}

// ---------------------------------------------------------------------------
// 8) Normalized Gaussian smoothing kernels per band.
// ---------------------------------------------------------------------------
__global__ void vox_gk(float* __restrict__ gk, int* __restrict__ gki) {
  const int band = blockIdx.x;
  const float smooth = exp2f((float)band * 0.5f);
  int ksz = (int)(128.0f / smooth);
  if (ksz < 3) ksz = 3;
  if ((ksz & 1) == 0) ksz += 1;
  const float sigma = (float)ksz / 6.0f;
  const int h = ksz / 2;
  __shared__ float sg[256];
  float v = 0.0f;
  if ((int)threadIdx.x < ksz) {
    const float d = (float)((int)threadIdx.x - h);
    v = __expf(-d * d / (2.0f * sigma * sigma));
  }
  sg[threadIdx.x] = v;
  __syncthreads();
  for (int off = 128; off > 0; off >>= 1) {
    if ((int)threadIdx.x < off) sg[threadIdx.x] += sg[threadIdx.x + off];
    __syncthreads();
  }
  const float inv = 1.0f / sg[0];
  if ((int)threadIdx.x < ksz) gk[band * 132 + threadIdx.x] = v * inv;
  if (threadIdx.x == 0) gki[band] = ksz;
}

// ---------------------------------------------------------------------------
// 9) 8-band shaped noise, LDS-tiled (256 outputs + 64 halo each side).
// ---------------------------------------------------------------------------
__global__ void vox_shape(const float* __restrict__ nz,
                          const float* __restrict__ bss,
                          const float* __restrict__ gk,
                          const int* __restrict__ gki,
                          float* __restrict__ shaped) {
  __shared__ float tilebuf[256 + 128];
  __shared__ float gks[8 * 132];
  __shared__ float wgt[8];
  __shared__ int   ks[8];
  const size_t blk = blockIdx.x;
  const int b  = (int)((blk * 256) / T_);
  const int t0 = (int)((blk * 256) % T_);
  const int tid = threadIdx.x;
  if (tid < 8) {
    wgt[tid] = bss[((size_t)b * T_ + T_ / 2) * 8 + tid];
    ks[tid]  = gki[tid];
  }
  for (int i = tid; i < 8 * 132; i += 256) gks[i] = gk[i];
  const int g0 = t0 - 64;
  for (int i = tid; i < 384; i += 256) {
    const int t = g0 + i;
    tilebuf[i] = (t >= 0 && t < T_) ? nz[(size_t)b * T_ + t] : 0.0f;
  }
  __syncthreads();
  float acc = 0.0f;
  for (int band = 0; band < 8; ++band) {
    const int k = ks[band];
    const int h = k >> 1;
    const float* gp = gks + band * 132;
    float s = 0.0f;
    for (int d = 0; d < k; ++d) s = __fmaf_rn(tilebuf[tid + 64 - h + d], gp[d], s);
    acc = __fmaf_rn(s, wgt[band], acc);
  }
  shaped[(size_t)b * T_ + t0 + tid] = acc;
}

// ---------------------------------------------------------------------------
// 10) Per-batch mean / std (ddof=1) of shaped noise.
// ---------------------------------------------------------------------------
__global__ void vox_stats(const float* __restrict__ sh, float* __restrict__ st) {
  const int b = blockIdx.x;
  const int tid = threadIdx.x;
  __shared__ float s1[1024];
  __shared__ float s2[1024];
  float a = 0.0f, q = 0.0f;
  for (int t = tid; t < T_; t += 1024) {
    const float v = sh[(size_t)b * T_ + t];
    a += v; q = __fmaf_rn(v, v, q);
  }
  s1[tid] = a; s2[tid] = q;
  __syncthreads();
  for (int off = 512; off > 0; off >>= 1) {
    if (tid < off) { s1[tid] += s1[tid + off]; s2[tid] += s2[tid + off]; }
    __syncthreads();
  }
  if (tid == 0) {
    const float mean = s1[0] / (float)T_;
    const float var  = (s2[0] - (float)T_ * mean * mean) / (float)(T_ - 1);
    st[b * 2 + 0] = mean;
    st[b * 2 + 1] = sqrtf(fmaxf(var, 0.0f)) + 1e-8f;
  }
}

// ---------------------------------------------------------------------------
// 11) Final mix: out = filtered*(1-br) + ((shaped-mean)/std)*br
// ---------------------------------------------------------------------------
__global__ void vox_mix(const float* __restrict__ filt,
                        const float* __restrict__ shaped,
                        const float* __restrict__ st,
                        const float* __restrict__ brp,
                        float* __restrict__ out) {
  const size_t idx = (size_t)blockIdx.x * blockDim.x + threadIdx.x;
  if (idx >= (size_t)B_ * T_) return;
  const int b = (int)(idx / T_);
  const float br = brp[idx];
  const float bn = (shaped[idx] - st[b * 2 + 0]) / st[b * 2 + 1];
  out[idx] = filt[idx] * (1.0f - br) + bn * br;
}

// ---------------------------------------------------------------------------
extern "C" void kernel_launch(void* const* d_in, const int* in_sizes, int n_in,
                              void* d_out, int out_size, void* d_ws, size_t ws_size,
                              hipStream_t stream) {
  const float* f0  = (const float*)d_in[0];
  const float* amp = (const float*)d_in[1];
  const float* vr  = (const float*)d_in[2];
  const float* vd  = (const float*)d_in[3];
  const float* ff  = (const float*)d_in[4];
  const float* fb  = (const float*)d_in[5];
  const float* fg  = (const float*)d_in[6];
  const float* oq  = (const float*)d_in[7];
  const float* pc  = (const float*)d_in[8]; (void)pc;  // blend is identity
  const float* br  = (const float*)d_in[9];
  const float* bss = (const float*)d_in[10];
  const float* ip  = (const float*)d_in[11];
  (void)in_sizes; (void)n_in; (void)out_size; (void)ws_size;

  float* out = (float*)d_out;   // [0,B*T): signal; [B*T,B*T+B): final phase

  char* w = (char*)d_ws;
  size_t off = 0;
  auto carve = [&](size_t bytes) -> char* {
    char* p = w + off;
    off += (bytes + 255) & ~(size_t)255;
    return p;
  };
  float*    fund   = (float*)   carve((size_t)B_ * T_ * sizeof(float));
  float*    f0v    = (float*)   carve((size_t)B_ * T_ * sizeof(float));
  float*    sig    = (float*)   carve((size_t)B_ * T_ * sizeof(float));
  float*    filt   = (float*)   carve((size_t)B_ * T_ * sizeof(float));
  float*    noise  = (float*)   carve((size_t)B_ * T_ * sizeof(float));
  float*    shaped = (float*)   carve((size_t)B_ * T_ * sizeof(float));
  float*    hbuf   = (float*)   carve((size_t)B_ * LH * sizeof(float));
  _Float16* xh     = (_Float16*)carve((size_t)B_ * PADLEN * sizeof(_Float16));
  _Float16* afrag  = (_Float16*)carve((size_t)B_ * SA_HALVES * sizeof(_Float16));
  float*    gk     = (float*)   carve(8 * 132 * sizeof(float));
  int*      gki    = (int*)     carve(8 * sizeof(int));
  float*    stats  = (float*)   carve(B_ * 2 * sizeof(float));

  const int bt_blocks  = (B_ * T_) / 256;               // 1000
  const int pad_blocks = (B_ * PADLEN + 255) / 256;

  vox_phase_scan<<<B_, 1024, 0, stream>>>(f0, vr, vd, ip, fund, f0v, out + (size_t)B_ * T_);
  vox_harmonics <<<bt_blocks, 256, 0, stream>>>(amp, f0v, fund, oq, sig);
  vox_hbuild    <<<B_, 256, 0, stream>>>(ff, fb, fg, hbuf);
  vox_pad16     <<<pad_blocks, 256, 0, stream>>>(sig, xh);
  vox_afrag     <<<B_ * NCHUNK, 32, 0, stream>>>(hbuf, afrag);
  vox_conv_wmma <<<B_ * BLK_PER_B, TPB_CONV * 32, 0, stream>>>(xh, afrag, filt);
  vox_noise     <<<bt_blocks, 256, 0, stream>>>(noise);
  vox_gk        <<<8, 256, 0, stream>>>(gk, gki);
  vox_shape     <<<bt_blocks, 256, 0, stream>>>(noise, bss, gk, gki, shaped);
  vox_stats     <<<B_, 1024, 0, stream>>>(shaped, stats);
  vox_mix       <<<bt_blocks, 256, 0, stream>>>(filt, shaped, stats, br, out);
}
